// QTAttB_21620865368131
// MI455X (gfx1250) — matrix operand order repaired and checked by reading, not compiled, not run on hardware
//
#include <hip/hip_runtime.h>

// ---------------------------------------------------------------------------
// QuadTree attention (QTAttB) for MI455X / gfx1250, wave32 + WMMA.
// B=4, C=128, NHEAD=8 -> d=16; coarse 32x32 (L=1024), fine 64x64.
// Coarse: dense attention per (b,h) on v_wmma_f32_16x16x32_bf16 with
// fragment-native bf16 operand layouts precomputed in workspace.
// Fine: per-(b,l,h) 4x32 gathered attention on VALU (gather-bound).
// ---------------------------------------------------------------------------

typedef __attribute__((ext_vector_type(16))) __bf16        v16bf;
typedef __attribute__((ext_vector_type(8)))  float         v8f;
typedef __attribute__((ext_vector_type(4)))  unsigned int  u32x4;
typedef __attribute__((ext_vector_type(2)))  unsigned int  u32x2;

#define NHEADS 8
#define DHEAD  16
#define LC     1024   // coarse positions (32*32)
#define LF     4096   // fine positions (64*64)

// Native conversions: let clang lower fptrunc/fpext to gfx1250 bf16 cvt ops
// (packed where possible) instead of manual integer rounding sequences.
static __device__ inline unsigned short f2bf_u(float f) {
  __bf16 h = (__bf16)f;
  return __builtin_bit_cast(unsigned short, h);
}
static __device__ inline unsigned pack2bf(float a, float b) {
  return (unsigned)f2bf_u(a) | ((unsigned)f2bf_u(b) << 16);
}
static __device__ inline float bf2f(unsigned short h) {
  return (float)__builtin_bit_cast(__bf16, h);
}

// ---------------------------------------------------------------------------
// Prep: fp32 coarse Q/K/V -> bf16 in WMMA-fragment-friendly layouts.
//   qbf,kbf : [b*8+h][pos][dim]  (Q pre-scaled by temp=0.25)
//   vbf     : [b*8+h][dim][pos]  (== input flattening, elementwise)
// One thread per output dword (2 bf16).  262144 threads.
// ---------------------------------------------------------------------------
__global__ __launch_bounds__(256) void qtatt_prep_kernel(
    const float* __restrict__ qc, const float* __restrict__ kc,
    const float* __restrict__ vc,
    unsigned* __restrict__ qbf, unsigned* __restrict__ kbf,
    unsigned* __restrict__ vbf)
{
  const int t = blockIdx.x * 256 + threadIdx.x;      // < 262144
  // V: layout-preserving elementwise convert (contiguous pair).
  vbf[t] = pack2bf(vc[2 * t], vc[2 * t + 1]);
  // Q/K: transpose channel-major -> [bh][pos][dim]; dword t = elements (2t,2t+1).
  const int dd  = (2 * t) & 15;                      // even
  const int pos = (t >> 3) & 1023;
  const int bh  = t >> 13;
  const size_t in0 = ((size_t)(bh * 16 + dd)) * LC + pos;
  qbf[t] = pack2bf(qc[in0] * 0.25f, qc[in0 + LC] * 0.25f);
  kbf[t] = pack2bf(kc[in0], kc[in0 + LC]);
}

// ---------------------------------------------------------------------------
// Kernel A: coarse dense attention + top-8 per (query,head).
// One wave per (b, h, 16-query row tile).  grid = 4*8*64 = 2048 x 32 threads.
// ---------------------------------------------------------------------------
__global__ __launch_bounds__(32) void qtatt_coarse_kernel(
    const unsigned short* __restrict__ qbf,
    const unsigned short* __restrict__ kbf,
    const unsigned short* __restrict__ vbf,
    float* __restrict__ msg0, int* __restrict__ topk_idx)
{
  __shared__ unsigned short sS[16 * LC];   // bf16 logits -> probs, 32 KB
  __shared__ float sMax[16];
  __shared__ float sSum[16];

  const int lane = threadIdx.x;            // 0..31
  const int bid  = blockIdx.x;
  const int mt = bid & 63;
  const int h  = (bid >> 6) & 7;
  const int b  = bid >> 9;
  const int m0 = mt * 16;
  const int bh = b * 8 + h;

  const unsigned short* qh = qbf + (size_t)bh * LC * DHEAD;
  const unsigned short* kh = kbf + (size_t)bh * LC * DHEAD;
  const unsigned short* vh = vbf + (size_t)bh * LC * DHEAD;

  // --- A fragment: Q tile (M=16, d=16 zero-padded to K=32).
  // 16-bit A 16x32 layout: lanes 0-15: M=lane, K=e / 16+e; lanes 16-31:
  // M=lane-16, K=8+e / 24+e.  One b128 load, high half zero.
  const int arow = lane & 15;
  const int koff = (lane >> 4) * 8;
  const u32x4 ql = *(const u32x4*)(qh + (m0 + arow) * DHEAD + koff);
  const u32x4 zz = {0u, 0u, 0u, 0u};
  const v16bf aQ = __builtin_bit_cast(
      v16bf, __builtin_shufflevector(ql, zz, 0, 1, 2, 3, 4, 5, 6, 7));

  // --- Pass 1: logits S = Q K^T (temp folded into Q) over 64 key tiles.
  const int mrow = (lane >> 4) * 8;
  const int ncol = lane & 15;
  for (int st = 0; st < 64; ++st) {
    const int s0 = st * 16;
    // B fragment (32x16): lanes 0-15: K-row=e (dims 0-7), lanes 16-31: K-row=8+e
    // (dims 8-15), N=lane&15 (key); elements 8-15 -> zero padding.  One 16B load.
    const u32x4 kl = *(const u32x4*)(kh + (s0 + (lane & 15)) * DHEAD + (lane >> 4) * 8);
    const v16bf bK = __builtin_bit_cast(
        v16bf, __builtin_shufflevector(kl, zz, 0, 1, 2, 3, 4, 5, 6, 7));
    if (lane < 16) {
      if (st + 1 < 64)
        __builtin_prefetch(kh + (s0 + 16 + lane) * DHEAD, 0, 1);
    }
    v8f c = {};
    c = __builtin_amdgcn_wmma_f32_16x16x32_bf16(false, aQ, false, bK,
                                                (short)0, c, false, false);
    // C layout: element r, lane n -> M = r + 8*(n>=16), N = n&15.
#pragma unroll
    for (int r = 0; r < 8; ++r)
      sS[(r + mrow) * LC + (s0 + ncol)] = f2bf_u(c[r]);
  }
  __syncthreads();

  // --- Top-8 indices + row max (softmax monotone -> select on logits).
  if (lane < 16) {
    const int row = lane;
    float tv[8]; int ti[8];
#pragma unroll
    for (int j = 0; j < 8; ++j) { tv[j] = -3.0e38f; ti[j] = 0; }
    const u32x2* prow2 = (const u32x2*)(sS + row * LC);
    for (int s = 0; s < LC; s += 4) {
      const u32x2 w4 = prow2[s >> 2];            // 4 bf16 logits
#pragma unroll
      for (int q4 = 0; q4 < 4; ++q4) {
        const unsigned dw = w4[q4 >> 1];
        float v = bf2f((unsigned short)((q4 & 1) ? (dw >> 16) : dw));
        if (v > tv[7]) {                         // rare: sorted insert
          int id = s + q4;
#pragma unroll
          for (int j = 0; j < 8; ++j) {
            bool gt = v > tv[j];
            float ov = tv[j]; int oi = ti[j];
            tv[j] = gt ? v : ov; ti[j] = gt ? id : oi;
            v = gt ? ov : v;     id = gt ? oi : id;
          }
        }
      }
    }
    sMax[row] = tv[0];
    int* op = topk_idx + ((size_t)((b * LC + m0 + row) * NHEADS + h)) * 8;
#pragma unroll
    for (int j = 0; j < 8; ++j) op[j] = ti[j];
  }
  __syncthreads();

  // --- exp(s - max) in place, packed pairs; row sums via wave32 shuffles.
  for (int r = 0; r < 16; ++r) {
    const float m = sMax[r];
    float acc = 0.0f;
    unsigned* rowp = (unsigned*)(sS + r * LC);
    for (int s1 = lane; s1 < (LC >> 1); s1 += 32) {
      const unsigned pr = rowp[s1];
      const float p0 = __expf(bf2f((unsigned short)pr) - m);
      const float p1 = __expf(bf2f((unsigned short)(pr >> 16)) - m);
      rowp[s1] = pack2bf(p0, p1);
      acc += p0 + p1;
    }
#pragma unroll
    for (int off = 16; off > 0; off >>= 1) acc += __shfl_xor(acc, off, 32);
    if (lane == 0) sSum[r] = acc;
  }
  __syncthreads();

  // --- Pass 2: O = P (16x1024 bf16) * V (1024x16), K=32 keys per WMMA.
  v8f o = {};
  const int prw  = lane & 15;
  const int pk   = (lane >> 4) * 8;
  const int vdim = lane & 15;
  const int vk   = (lane >> 4) * 16;
  for (int kt = 0; kt < 32; ++kt) {
    const int kb = kt * 32;
    // A fragment: two ds_load_b128 (keys kb+2*pk..+7 and kb+2*pk+8..+15).
    const u32x4 t0 = *(const u32x4*)(sS + prw * LC + kb + 2 * pk);
    const u32x4 t1 = *(const u32x4*)(sS + prw * LC + kb + 2 * pk + 8);
    const v16bf aP = __builtin_bit_cast(
        v16bf, __builtin_shufflevector(t0, t1, 0, 1, 2, 3, 4, 5, 6, 7));
    // B fragment: 16 consecutive keys at fixed dim -> one 32B load.
    const v16bf bV = *(const v16bf*)(vh + vdim * LC + kb + vk);
    if (kt + 1 < 32)
      __builtin_prefetch(vh + vdim * LC + kb + 32 + vk, 0, 1);
    o = __builtin_amdgcn_wmma_f32_16x16x32_bf16(false, aP, false, bV,
                                                (short)0, o, false, false);
  }

  // --- Normalize and write msg0 [b, Lc, h, d].
  const int orow = (lane >> 4) * 8;
  const int odim = lane & 15;
#pragma unroll
  for (int r = 0; r < 8; ++r) {
    const int row = r + orow;
    const float val = o[r] / sSum[row];
    msg0[((size_t)((b * LC + m0 + row) * NHEADS + h)) * DHEAD + odim] = val;
  }
}

// ---------------------------------------------------------------------------
// Kernel B: fine-level gathered attention + combine.
// One wave per (b, l, h); lane = one of 32 candidates (top-8 x 2x2).
// ---------------------------------------------------------------------------
__global__ __launch_bounds__(256) void qtatt_fine_kernel(
    const float* __restrict__ qf, const float* __restrict__ kf,
    const float* __restrict__ vf, const float* __restrict__ wt,
    const float* __restrict__ msg0, const int* __restrict__ topk_idx,
    float* __restrict__ out)
{
  __shared__ float sQ[8][4][16];
  __shared__ float sV[8][32][16];
  __shared__ float sP[8][4][32];

  const int lane = threadIdx.x & 31;
  const int w    = threadIdx.x >> 5;
  const int inst = blockIdx.x * 8 + w;     // ((b*1024)+l)*8 + h
  const int h = inst & 7;
  const int l = (inst >> 3) & 1023;
  const int b = inst >> 13;

  const int i2 = l >> 5, j2 = l & 31;
  const size_t chanBase = (size_t)(b * 128 + h * DHEAD) * LF;
  const float* qch = qf + chanBase;
  const float* kch = kf + chanBase;
  const float* vch = vf + chanBase;

  // Candidate fine position: coarse topk j = lane/4, 2x2 sub = lane&3.
  const int j   = lane >> 2;
  const int sub = lane & 3;
  const int cpos = topk_idx[((size_t)((b * LC + l) * NHEADS + h)) * 8 + j];
  const int fy = 2 * (cpos >> 5) + (sub >> 1);
  const int fx = 2 * (cpos & 31) + (sub & 1);
  const int fpos = fy * 64 + fx;

  float kreg[16];
#pragma unroll
  for (int dd = 0; dd < 16; ++dd) {
    kreg[dd] = kch[(size_t)dd * LF + fpos];
    sV[w][lane][dd] = vch[(size_t)dd * LF + fpos];
  }
#pragma unroll
  for (int half = 0; half < 2; ++half) {
    const int t = lane + half * 32;          // 0..63
    const int q = t >> 4, dd = t & 15;
    const int qpos = (2 * i2 + (q >> 1)) * 64 + (2 * j2 + (q & 1));
    sQ[w][q][dd] = qch[(size_t)dd * LF + qpos] * 0.25f;
  }
  __syncthreads();

  float S[4];
#pragma unroll
  for (int q = 0; q < 4; ++q) {
    float s = 0.0f;
#pragma unroll
    for (int dd = 0; dd < 16; ++dd) s += sQ[w][q][dd] * kreg[dd];
    S[q] = s;
  }
#pragma unroll
  for (int q = 0; q < 4; ++q) {
    float m = S[q];
#pragma unroll
    for (int off = 16; off > 0; off >>= 1) m = fmaxf(m, __shfl_xor(m, off, 32));
    const float p = __expf(S[q] - m);
    float ssum = p;
#pragma unroll
    for (int off = 16; off > 0; off >>= 1) ssum += __shfl_xor(ssum, off, 32);
    sP[w][q][lane] = p / ssum;
  }
  __syncthreads();

  const float mw = fmaxf(wt[0], wt[1]);
  const float e0 = __expf(wt[0] - mw), e1 = __expf(wt[1] - mw);
  const float w0 = e0 / (e0 + e1), w1 = e1 / (e0 + e1);

#pragma unroll
  for (int half = 0; half < 2; ++half) {
    const int t = lane + half * 32;          // 0..63 -> (q, dd)
    const int q = t >> 4, dd = t & 15;
    float acc = 0.0f;
#pragma unroll
    for (int s = 0; s < 32; ++s) acc += sP[w][q][s] * sV[w][s][dd];
    const float m0v = msg0[((size_t)((b * LC + l) * NHEADS + h)) * DHEAD + dd];
    const int opos = (2 * i2 + (q >> 1)) * 64 + (2 * j2 + (q & 1));
    out[((size_t)((b * LF + opos) * NHEADS + h)) * DHEAD + dd] =
        m0v * w0 + acc * w1;
  }
}

// ---------------------------------------------------------------------------
extern "C" void kernel_launch(void* const* d_in, const int* in_sizes, int n_in,
                              void* d_out, int out_size, void* d_ws, size_t ws_size,
                              hipStream_t stream) {
  (void)in_sizes; (void)n_in; (void)out_size; (void)ws_size;
  const float* q_fine   = (const float*)d_in[0];
  const float* q_coarse = (const float*)d_in[1];
  const float* k_fine   = (const float*)d_in[2];
  const float* k_coarse = (const float*)d_in[3];
  const float* v_fine   = (const float*)d_in[4];
  const float* v_coarse = (const float*)d_in[5];
  const float* weight   = (const float*)d_in[6];
  float* out = (float*)d_out;

  // Workspace layout (6 MB):
  //   msg0 [4,1024,8,16] f32   : 2 MB @ 0
  //   tidx [4,1024,8,8]  i32   : 1 MB @ 2 MB
  //   qbf  [4*8,1024,16] bf16  : 1 MB @ 3 MB  (temp-scaled)
  //   kbf  [4*8,1024,16] bf16  : 1 MB @ 4 MB
  //   vbf  [4*8,16,1024] bf16  : 1 MB @ 5 MB
  char* ws = (char*)d_ws;
  float*          msg0 = (float*)ws;
  int*            tidx = (int*)(ws + (2u << 20));
  unsigned short* qbf  = (unsigned short*)(ws + (3u << 20));
  unsigned short* kbf  = (unsigned short*)(ws + (4u << 20));
  unsigned short* vbf  = (unsigned short*)(ws + (5u << 20));

  qtatt_prep_kernel<<<1024, 256, 0, stream>>>(q_coarse, k_coarse, v_coarse,
                                              (unsigned*)qbf, (unsigned*)kbf,
                                              (unsigned*)vbf);
  qtatt_coarse_kernel<<<4 * 8 * 64, 32, 0, stream>>>(qbf, kbf, vbf, msg0, tidx);
  qtatt_fine_kernel<<<4096, 256, 0, stream>>>(q_fine, k_fine, v_fine, weight,
                                              msg0, tidx, out);
}
